// TransformerEncoder_44152263803591
// MI455X (gfx1250) — matrix operand order repaired and checked by reading, not compiled, not run on hardware
//
#include <hip/hip_runtime.h>
#include <stdint.h>

// Problem constants (fixed by the reference)
#define Bsz   2
#define Lsz   2048
#define Dsz   1024
#define Hn    16
#define Ehd   64
#define Mrows (Bsz * Lsz)   // 4096

typedef _Float16 v8h  __attribute__((ext_vector_type(8)));
typedef _Float16 v16h __attribute__((ext_vector_type(16)));
typedef float    v8f  __attribute__((ext_vector_type(8)));

static __device__ __forceinline__ v16h comb16(v8h lo, v8h hi) {
  return __builtin_shufflevector(lo, hi, 0,1,2,3,4,5,6,7,8,9,10,11,12,13,14,15);
}
static __device__ __forceinline__ v8f wmma16(v16h a, v16h b, v8f c) {
  // D = A(16x32 f16) * B(32x16 f16) + C(16x16 f32)
  return __builtin_amdgcn_wmma_f32_16x16x32_f16(false, a, false, b, (short)0, c, false, false);
}
// 16B global -> LDS direct copy, tracked by ASYNCcnt (CDNA5 §15.18.3 op 98)
static __device__ __forceinline__ void async_copy16(unsigned lds_addr, const void* gaddr) {
  asm volatile("global_load_async_to_lds_b128 %0, %1, off"
               :: "v"(lds_addr), "v"(gaddr) : "memory");
}
static __device__ __forceinline__ void wait_async0() {
  asm volatile("s_wait_asynccnt 0x0" ::: "memory");
}
static __device__ __forceinline__ unsigned lds_lo32(const void* p) {
  return (unsigned)(uintptr_t)p;   // flat LDS address: addr[31:0] is the LDS offset
}

// ---------------------------------------------------------------- cvt f32->f16
__global__ __launch_bounds__(256) void cvt_f32_f16(const float* __restrict__ in,
                                                   _Float16* __restrict__ out, int n) {
  int i = (blockIdx.x * blockDim.x + threadIdx.x) * 4;
  if (i + 3 < n) {
    float4 v = *(const float4*)(in + i);
    out[i + 0] = (_Float16)v.x;
    out[i + 1] = (_Float16)v.y;
    out[i + 2] = (_Float16)v.z;
    out[i + 3] = (_Float16)v.w;
  } else {
    for (int j = i; j < n; ++j) out[j] = (_Float16)in[j];
  }
}

// ---------------------------------------------------------------- WMMA GEMM
// C[M,N] (f32) = A[M,K](f16, row-major) @ W[N,K](f16)^T  (i.e. x @ W^T)
// Double-buffered LDS, next K-tile staged with global_load_async_to_lds_b128.
__global__ __launch_bounds__(256) void gemm_f16(
    const _Float16* __restrict__ A, const _Float16* __restrict__ W,
    float* __restrict__ Cf, _Float16* __restrict__ Cheads,
    const float* __restrict__ bias, int M, int N, int K) {
  __shared__ __align__(16) _Float16 sA[2][128][40];   // 32 K-halfs + 8 pad
  __shared__ __align__(16) _Float16 sB[2][64][40];

  const int tid  = threadIdx.x;
  const int wave = tid >> 5, lane = tid & 31;
  const int g = lane >> 4, ln = lane & 15;
  const int wm = wave >> 1, wn = wave & 1;            // 4x2 wave grid, each wave 32x32
  const int mbase = blockIdx.y * 128, nbase = blockIdx.x * 64;

  // 16B staging chunks: A = 512 chunks (2/thread), B = 256 chunks (1/thread)
  const int arow0 = tid >> 2;                  // rows 0..63
  const int arow1 = 64 + (tid >> 2);           // rows 64..127
  const int cko   = (tid & 3) * 8;             // half offset within row
  const _Float16* Ab = A + (size_t)mbase * K;
  const _Float16* Wb = W + (size_t)nbase * K;

  auto issue = [&](int k0, int buf) {
    async_copy16(lds_lo32(&sA[buf][arow0][cko]), Ab + (size_t)arow0 * K + k0 + cko);
    async_copy16(lds_lo32(&sA[buf][arow1][cko]), Ab + (size_t)arow1 * K + k0 + cko);
    async_copy16(lds_lo32(&sB[buf][arow0][cko]), Wb + (size_t)arow0 * K + k0 + cko);
  };

  v8f zero = {};
  v8f acc[2][2];
#pragma unroll
  for (int i = 0; i < 2; ++i)
#pragma unroll
    for (int j = 0; j < 2; ++j) acc[i][j] = zero;

  issue(0, 0);
  for (int k0 = 0; k0 < K; k0 += 32) {
    const int buf = (k0 >> 5) & 1;
    wait_async0();        // my async chunks for `buf` have landed
    __syncthreads();      // everyone's have
    if (k0 + 32 < K) {
      issue(k0 + 32, buf ^ 1);    // overlap next tile with this tile's WMMAs
      __builtin_prefetch(Ab + (size_t)arow0 * K + k0 + 64 + cko, 0, 3);
    }

    v16h af[2], bf[2];
#pragma unroll
    for (int mt = 0; mt < 2; ++mt) {
      int r = wm * 32 + mt * 16 + ln;
      af[mt] = comb16(*(const v8h*)&sA[buf][r][8 * g], *(const v8h*)&sA[buf][r][16 + 8 * g]);
    }
#pragma unroll
    for (int nt = 0; nt < 2; ++nt) {
      int r = wn * 32 + nt * 16 + ln;
      bf[nt] = comb16(*(const v8h*)&sB[buf][r][8 * g], *(const v8h*)&sB[buf][r][16 + 8 * g]);
    }
#pragma unroll
    for (int mt = 0; mt < 2; ++mt)
#pragma unroll
      for (int nt = 0; nt < 2; ++nt)
        acc[mt][nt] = wmma16(af[mt], bf[nt], acc[mt][nt]);
  }

#pragma unroll
  for (int mt = 0; mt < 2; ++mt)
#pragma unroll
    for (int nt = 0; nt < 2; ++nt)
#pragma unroll
      for (int r = 0; r < 8; ++r) {
        int m = mbase + wm * 32 + mt * 16 + r + 8 * g;
        int n = nbase + wn * 32 + nt * 16 + ln;
        float v = acc[mt][nt][r];
        if (Cf) Cf[(size_t)m * N + n] = bias ? (v + bias[n]) : v;
        if (Cheads) {
          int b = m >> 11, l = m & (Lsz - 1);     // L = 2048
          int h = n >> 6, e = n & (Ehd - 1);      // E = 64
          Cheads[(((size_t)b * Hn + h) * Lsz + l) * Ehd + e] = (_Float16)v;
        }
      }
}

// ---------------------------------------------------------------- flash attention
// grid = (B*H, L/64), block = 128 (4 waves, 16 q-rows each). Q/K/V in [B,H,L,E] f16.
// ctx out: f16 [B, L, H*E]. K tile: async double-buffered. V tile: register-staged
// transpose, loads issued early / stores committed after compute.
__global__ __launch_bounds__(128) void flash_attn(
    const _Float16* __restrict__ Q, const _Float16* __restrict__ Kh,
    const _Float16* __restrict__ V, _Float16* __restrict__ ctx) {
  __shared__ __align__(16) _Float16 sK[2][64][72];     // [key][e]  (K^T B-frags)
  __shared__ __align__(16) _Float16 sV[2][64][72];     // [e][key]  (V   B-frags)
  __shared__ __align__(16) _Float16 sP[4][16][72];     // per-wave P tile

  const int tid = threadIdx.x;
  const int wave = tid >> 5, lane = tid & 31;
  const int g = lane >> 4, ln = lane & 15;
  const int bh = blockIdx.x;                 // b*H + h
  const int qb = blockIdx.y * 64;
  const size_t headbase = (size_t)bh * Lsz * Ehd;

  // 16B staging chunks: 64 rows x 4 chunks = 256 chunks, 2 per thread
  const int row0 = tid >> 2;            // rows 0..31
  const int row1 = 32 + (tid >> 2);     // rows 32..63
  const int cko  = (tid & 3) * 8;

  auto issueK = [&](int kb, int buf) {
    async_copy16(lds_lo32(&sK[buf][row0][cko]), Kh + headbase + (size_t)(kb + row0) * Ehd + cko);
    async_copy16(lds_lo32(&sK[buf][row1][cko]), Kh + headbase + (size_t)(kb + row1) * Ehd + cko);
  };

  // Q A-fragments (registers for whole kernel)
  const _Float16* qp = Q + headbase + (size_t)(qb + wave * 16 + ln) * Ehd;
  v16h qf[2];
#pragma unroll
  for (int f = 0; f < 2; ++f)
    qf[f] = comb16(*(const v8h*)&qp[f * 32 + 8 * g], *(const v8h*)&qp[f * 32 + 16 + 8 * g]);

  v8f zero = {};
  v8f cacc[4];
#pragma unroll
  for (int t = 0; t < 4; ++t) cacc[t] = zero;
  float rmax[8], rsum[8];
#pragma unroll
  for (int r = 0; r < 8; ++r) { rmax[r] = -1e30f; rsum[r] = 0.0f; }

  // prologue: stage block 0
  issueK(0, 0);
  {
    v8h t0 = *(const v8h*)&V[headbase + (size_t)row0 * Ehd + cko];
    v8h t1 = *(const v8h*)&V[headbase + (size_t)row1 * Ehd + cko];
#pragma unroll
    for (int i = 0; i < 8; ++i) { sV[0][cko + i][row0] = t0[i]; sV[0][cko + i][row1] = t1[i]; }
  }

  for (int kb = 0; kb < Lsz; kb += 64) {
    const int buf = (kb >> 6) & 1, nb = buf ^ 1;
    const bool more = (kb + 64) < Lsz;
    wait_async0();
    __syncthreads();

    v8h vr0, vr1;
    if (more) {   // start next block: async K copy + V loads into registers
      issueK(kb + 64, nb);
      vr0 = *(const v8h*)&V[headbase + (size_t)(kb + 64 + row0) * Ehd + cko];
      vr1 = *(const v8h*)&V[headbase + (size_t)(kb + 64 + row1) * Ehd + cko];
    }

    // S = Q K^T * (1/sqrt(E))
    float s[4][8];
#pragma unroll
    for (int t = 0; t < 4; ++t) {
      v8f c0 = zero;
      int key = 16 * t + ln;
#pragma unroll
      for (int f = 0; f < 2; ++f) {
        v16h bfr = comb16(*(const v8h*)&sK[buf][key][f * 32 + 8 * g],
                          *(const v8h*)&sK[buf][key][f * 32 + 16 + 8 * g]);
        c0 = wmma16(qf[f], bfr, c0);
      }
#pragma unroll
      for (int r = 0; r < 8; ++r) s[t][r] = c0[r] * 0.125f;
    }

    // online softmax (rows live within a 16-lane half-wave)
#pragma unroll
    for (int r = 0; r < 8; ++r) {
      float mx = s[0][r];
#pragma unroll
      for (int t = 1; t < 4; ++t) mx = fmaxf(mx, s[t][r]);
#pragma unroll
      for (int off = 8; off >= 1; off >>= 1) mx = fmaxf(mx, __shfl_xor(mx, off, 16));
      float nm = fmaxf(rmax[r], mx);
      float corr = __expf(rmax[r] - nm);
      rmax[r] = nm;
      float ps = 0.0f;
#pragma unroll
      for (int t = 0; t < 4; ++t) { float p = __expf(s[t][r] - nm); s[t][r] = p; ps += p; }
#pragma unroll
      for (int off = 8; off >= 1; off >>= 1) ps += __shfl_xor(ps, off, 16);
      rsum[r] = rsum[r] * corr + ps;
#pragma unroll
      for (int t = 0; t < 4; ++t) cacc[t][r] *= corr;
    }

    // P (D-layout) -> per-wave LDS tile -> A-fragments
#pragma unroll
    for (int t = 0; t < 4; ++t)
#pragma unroll
      for (int r = 0; r < 8; ++r)
        sP[wave][r + 8 * g][16 * t + ln] = (_Float16)s[t][r];

    v16h pf[2];
#pragma unroll
    for (int f = 0; f < 2; ++f)
      pf[f] = comb16(*(const v8h*)&sP[wave][ln][f * 32 + 8 * g],
                     *(const v8h*)&sP[wave][ln][f * 32 + 16 + 8 * g]);

    // ctx += P @ V
#pragma unroll
    for (int t = 0; t < 4; ++t) {
      int e0 = 16 * t + ln;
#pragma unroll
      for (int f = 0; f < 2; ++f) {
        v16h bfr = comb16(*(const v8h*)&sV[buf][e0][f * 32 + 8 * g],
                          *(const v8h*)&sV[buf][e0][f * 32 + 16 + 8 * g]);
        cacc[t] = wmma16(pf[f], bfr, cacc[t]);
      }
    }

    if (more) {   // commit V transpose after compute; barrier next iter orders it
#pragma unroll
      for (int i = 0; i < 8; ++i) { sV[nb][cko + i][row0] = vr0[i]; sV[nb][cko + i][row1] = vr1[i]; }
    }
  }

  // write ctx [B, L, H*E], normalized by running sum
  const int b = bh / Hn, h = bh % Hn;
#pragma unroll
  for (int t = 0; t < 4; ++t)
#pragma unroll
    for (int r = 0; r < 8; ++r) {
      int m = qb + wave * 16 + r + 8 * g;
      float v = cacc[t][r] / rsum[r];
      ctx[((size_t)b * Lsz + m) * (Hn * Ehd) + h * Ehd + 16 * t + ln] = (_Float16)v;
    }
}

// ---------------------------------------------------------------- residual + LayerNorm
__global__ __launch_bounds__(256) void add_ln(const float* __restrict__ a,
                                              const float* __restrict__ b,
                                              float* __restrict__ out,
                                              _Float16* __restrict__ outh) {
  __shared__ float rs[8], rq[8];
  const int row = blockIdx.x, tid = threadIdx.x;
  const size_t base = (size_t)row * Dsz + tid * 4;
  float4 va = *(const float4*)(a + base);
  float4 vb = *(const float4*)(b + base);
  float x0 = va.x + vb.x, x1 = va.y + vb.y, x2 = va.z + vb.z, x3 = va.w + vb.w;
  float s = x0 + x1 + x2 + x3;
  float q = x0 * x0 + x1 * x1 + x2 * x2 + x3 * x3;
#pragma unroll
  for (int off = 16; off >= 1; off >>= 1) { s += __shfl_xor(s, off, 32); q += __shfl_xor(q, off, 32); }
  const int wv = tid >> 5, ln = tid & 31;
  if (ln == 0) { rs[wv] = s; rq[wv] = q; }
  __syncthreads();
  s = 0.0f; q = 0.0f;
#pragma unroll
  for (int w = 0; w < 8; ++w) { s += rs[w]; q += rq[w]; }
  float mean = s * (1.0f / Dsz);
  float var = q * (1.0f / Dsz) - mean * mean;
  float inv = rsqrtf(var + 1e-5f);
  float y0 = (x0 - mean) * inv, y1 = (x1 - mean) * inv;
  float y2 = (x2 - mean) * inv, y3 = (x3 - mean) * inv;
  float4 vo; vo.x = y0; vo.y = y1; vo.z = y2; vo.w = y3;
  *(float4*)(out + base) = vo;
  if (outh) {
    _Float16* hp = outh + base;
    hp[0] = (_Float16)y0; hp[1] = (_Float16)y1; hp[2] = (_Float16)y2; hp[3] = (_Float16)y3;
  }
}

// ---------------------------------------------------------------- host
extern "C" void kernel_launch(void* const* d_in, const int* in_sizes, int n_in,
                              void* d_out, int out_size, void* d_ws, size_t ws_size,
                              hipStream_t stream) {
  (void)in_sizes; (void)n_in; (void)out_size; (void)ws_size;
  const float* query = (const float*)d_in[0];
  const float* key_  = (const float*)d_in[1];
  const float* value = (const float*)d_in[2];
  const float* W_Q = (const float*)d_in[3];
  const float* W_K = (const float*)d_in[4];
  const float* W_V = (const float*)d_in[5];
  const float* W_O = (const float*)d_in[6];
  const float* fcw = (const float*)d_in[7];
  const float* fcb = (const float*)d_in[8];
  float* out = (float*)d_out;

  char* ws = (char*)d_ws;
  const size_t MD2 = (size_t)Mrows * Dsz * 2;   // f16 activation buffer
  const size_t WD2 = (size_t)Dsz * Dsz * 2;     // f16 weight buffer
  const size_t MD4 = (size_t)Mrows * Dsz * 4;   // f32 activation buffer
  size_t off = 0;
  _Float16* Qi  = (_Float16*)(ws + off); off += MD2;
  _Float16* Ki  = (_Float16*)(ws + off); off += MD2;
  _Float16* Vi  = (_Float16*)(ws + off); off += MD2;
  _Float16* WQh = (_Float16*)(ws + off); off += WD2;
  _Float16* WKh = (_Float16*)(ws + off); off += WD2;
  _Float16* WVh = (_Float16*)(ws + off); off += WD2;
  _Float16* WOh = (_Float16*)(ws + off); off += WD2;
  _Float16* FWh = (_Float16*)(ws + off); off += WD2;
  _Float16* qh  = (_Float16*)(ws + off); off += MD2;   // [B,H,L,E]
  _Float16* kh  = (_Float16*)(ws + off); off += MD2;
  _Float16* vh  = (_Float16*)(ws + off); off += MD2;
  _Float16* ctxh = (_Float16*)(ws + off); off += MD2;  // [B,L,H*E]
  float*    g32 = (float*)(ws + off); off += MD4;      // gemm f32 out (attn_out, then fc_out)
  float*    xf  = (float*)(ws + off); off += MD4;      // x = LN(attn_out + query)
  _Float16* xh  = (_Float16*)(ws + off); off += MD2;

  const int nMD = Mrows * Dsz;   // 4194304
  const int nWD = Dsz * Dsz;     // 1048576
  dim3 cvb(256);
  dim3 cvgA((nMD / 4 + 255) / 256), cvgW((nWD / 4 + 255) / 256);
  cvt_f32_f16<<<cvgA, cvb, 0, stream>>>(query, Qi, nMD);
  cvt_f32_f16<<<cvgA, cvb, 0, stream>>>(key_,  Ki, nMD);
  cvt_f32_f16<<<cvgA, cvb, 0, stream>>>(value, Vi, nMD);
  cvt_f32_f16<<<cvgW, cvb, 0, stream>>>(W_Q, WQh, nWD);
  cvt_f32_f16<<<cvgW, cvb, 0, stream>>>(W_K, WKh, nWD);
  cvt_f32_f16<<<cvgW, cvb, 0, stream>>>(W_V, WVh, nWD);
  cvt_f32_f16<<<cvgW, cvb, 0, stream>>>(W_O, WOh, nWD);
  cvt_f32_f16<<<cvgW, cvb, 0, stream>>>(fcw, FWh, nWD);

  dim3 gg(Dsz / 64, Mrows / 128);   // (16, 32)
  dim3 gb(256);
  gemm_f16<<<gg, gb, 0, stream>>>(Qi, WQh, nullptr, qh, nullptr, Mrows, Dsz, Dsz);
  gemm_f16<<<gg, gb, 0, stream>>>(Ki, WKh, nullptr, kh, nullptr, Mrows, Dsz, Dsz);
  gemm_f16<<<gg, gb, 0, stream>>>(Vi, WVh, nullptr, vh, nullptr, Mrows, Dsz, Dsz);
  flash_attn<<<dim3(Bsz * Hn, Lsz / 64), 128, 0, stream>>>(qh, kh, vh, ctxh);
  gemm_f16<<<gg, gb, 0, stream>>>(ctxh, WOh, g32, nullptr, nullptr, Mrows, Dsz, Dsz);
  add_ln<<<Mrows, 256, 0, stream>>>(g32, query, xf, xh);
  gemm_f16<<<gg, gb, 0, stream>>>(xh, FWh, g32, nullptr, fcb, Mrows, Dsz, Dsz);
  add_ln<<<Mrows, 256, 0, stream>>>(g32, xf, out, nullptr);
}
